// RQSpline_15169824489929
// MI455X (gfx1250) — matrix operand order repaired and checked by reading, not compiled, not run on hardware
//
#include <hip/hip_runtime.h>
#include <hip/hip_bf16.h>

// ---------------------------------------------------------------------------
// RQ-spline coupling-flow log_prob for MI455X (gfx1250, wave32, WMMA).
// All four per-layer GEMMs run as v_wmma_f32_16x16x32_bf16 tiles. Wout columns
// are padded to 32/feature in LDS so the Wout GEMM fuses with the spline in
// 2-feature chunks -> per-wave LDS ~9 KB -> 8 waves/block, 16 waves/WGP
// (4 per SIMD32) so spline VALU overlaps other waves' WMMA work.
// ---------------------------------------------------------------------------

typedef __bf16 bf16_t;
typedef __attribute__((ext_vector_type(16))) __bf16 v16bf;
typedef __attribute__((ext_vector_type(8)))  __bf16 v8bf;
typedef __attribute__((ext_vector_type(8)))  float  v8f;

#define WAVES    8
#define NTHREADS (WAVES * 32)

static constexpr int   FDIM  = 16;
static constexpr int   KBINS = 8;
static constexpr int   NLAY  = 8;
static constexpr int   NBP   = 3 * KBINS + 1;     // 25
static constexpr int   POUT  = FDIM * NBP;        // 400
static constexpr int   PPAD  = 32;                // padded params per feature
static constexpr int   POUTP = FDIM * PPAD;       // 512 padded columns
static constexpr int   CSTR  = 68;                // chunk row stride (floats)
static constexpr float RMINV = -10.0f, RMAXV = 10.0f;
static constexpr float MIN_BIN = 1e-4f, MIN_SLOPE = 1e-4f;
static constexpr float WSCALE  = (RMAXV - RMINV) - KBINS * MIN_BIN; // 19.9992
static constexpr float SOFT_OFF = 0.54116654f;     // log(expm1(1 - 1e-4))
static constexpr float BASE_C   = -14.7030165f;    // -0.5*F*log(2*pi)

struct SM {
  bf16_t w0t[16][32];             //  1 KB   [n][k], k 16..31 zero
  bf16_t w1t[64][32];             //  4 KB   [n][k], k 16..31 zero
  bf16_t w2t[64][64];             //  8 KB   [n][k]
  bf16_t woutt[POUTP][64];        // 64 KB   [f*32+j][k], j>=25 zero
  float  b0v[16], b1v[64], b2v[64], boutv[POUTP];
  float  sclv[16], shfv[16], lscv[16];
  float  zbuf[WAVES][16][17];     // per-wave z tile, padded stride
  bf16_t hidA[WAVES][16][64];     // activation ping
  bf16_t hidB[WAVES][16][64];     // activation pong
  float  pchunk[WAVES][16][CSTR]; // 2-feature spline-param chunk (f32)
};                                // total ~154 KB

#define LDS_FENCE() asm volatile("s_wait_dscnt 0" ::: "memory")

__device__ __forceinline__ v8f wmma_bf16(v16bf a, v16bf b, v8f c) {
  return __builtin_amdgcn_wmma_f32_16x16x32_bf16(false, a, false, b,
                                                 (short)0, c, false, false);
}

// A fragment (16x32 bf16, row-major, stride in elements).
// lane = 16*g + m; elems 0..7 -> K = kb+8g+j, 8..15 -> K = kb+16+8g+(j-8)
__device__ __forceinline__ v16bf load_afrag(const bf16_t* base, int stride,
                                            int m, int g, int kb) {
  const bf16_t* row = base + m * stride;
  v8bf lo = *(const v8bf*)(row + kb + 8 * g);
  v8bf hi = *(const v8bf*)(row + kb + 16 + 8 * g);
  v16bf r;
#pragma unroll
  for (int j = 0; j < 8; ++j) { r[j] = lo[j]; r[8 + j] = hi[j]; }
  return r;
}

// B fragment (32x16 bf16) from transposed weights [n][k]:
// lane = 16*g + n(%16); elems 0..7 -> K = kb+8g+j, 8..15 -> K = kb+16+8g+(j-8)
__device__ __forceinline__ v16bf load_bfrag(const bf16_t* baseT, int stride,
                                            int n, int g, int kb) {
  const bf16_t* row = baseT + n * stride;
  v8bf lo = *(const v8bf*)(row + kb + 8 * g);
  v8bf hi = *(const v8bf*)(row + kb + 16 + 8 * g);
  v16bf r;
#pragma unroll
  for (int j = 0; j < 8; ++j) { r[j] = lo[j]; r[8 + j] = hi[j]; }
  return r;
}

__device__ __forceinline__ float tanh_fast(float x) {
  x = fminf(fmaxf(x, -15.f), 15.f);
  float e = __expf(2.f * x);
  return (e - 1.f) / (e + 1.f);
}

// distrax RationalQuadraticSpline forward_and_log_det, one (row,feature) cell.
__device__ __forceinline__ void rq_spline(const float* __restrict__ p, float x,
                                          float& y, float& ld) {
  float uw[KBINS], uh[KBINS], ud[KBINS + 1];
#pragma unroll
  for (int j = 0; j < KBINS; ++j) uw[j] = p[j];
#pragma unroll
  for (int j = 0; j < KBINS; ++j) uh[j] = p[KBINS + j];
#pragma unroll
  for (int j = 0; j <= KBINS; ++j) ud[j] = p[2 * KBINS + j];

  float mw = uw[0], mh = uh[0];
#pragma unroll
  for (int j = 1; j < KBINS; ++j) { mw = fmaxf(mw, uw[j]); mh = fmaxf(mh, uh[j]); }
  float ew[KBINS], eh[KBINS], sw = 0.f, sh = 0.f;
#pragma unroll
  for (int j = 0; j < KBINS; ++j) {
    ew[j] = __expf(uw[j] - mw); sw += ew[j];
    eh[j] = __expf(uh[j] - mh); sh += eh[j];
  }
  float iw = WSCALE / sw, ih = WSCALE / sh;
  float xs[KBINS + 1], ys[KBINS + 1];
  xs[0] = RMINV; ys[0] = RMINV;
#pragma unroll
  for (int j = 0; j < KBINS; ++j) {
    xs[j + 1] = xs[j] + (ew[j] * iw + MIN_BIN);
    ys[j + 1] = ys[j] + (eh[j] * ih + MIN_BIN);
  }
  float dv[KBINS + 1];
#pragma unroll
  for (int j = 0; j <= KBINS; ++j) {
    float t = ud[j] + SOFT_OFF;
    dv[j] = fmaxf(t, 0.f) + __logf(1.f + __expf(-fabsf(t))) + MIN_SLOPE;
  }
  int idx = 0;
#pragma unroll
  for (int j = 1; j <= KBINS; ++j) idx += (x >= xs[j]) ? 1 : 0;
  idx = (idx > KBINS - 1) ? (KBINS - 1) : idx;
  float xk = xs[0], xk1 = xs[1], yk = ys[0], yk1 = ys[1], dk = dv[0], dk1 = dv[1];
#pragma unroll
  for (int j = 1; j < KBINS; ++j) {
    bool c = (idx == j);
    xk  = c ? xs[j]     : xk;   xk1 = c ? xs[j + 1] : xk1;
    yk  = c ? ys[j]     : yk;   yk1 = c ? ys[j + 1] : yk1;
    dk  = c ? dv[j]     : dk;   dk1 = c ? dv[j + 1] : dk1;
  }
  float bw = xk1 - xk, bh = yk1 - yk;
  float s  = bh / bw;
  float zt = fminf(fmaxf((x - xk) / bw, 0.f), 1.f);
  float z1 = 1.f - zt;
  float den = s + (dk1 + dk - 2.f * s) * zt * z1;
  float yy  = yk + bh * (s * zt * zt + dk * zt * z1) / den;
  float num = dk1 * zt * zt + 2.f * s * zt * z1 + dk * z1 * z1;
  float l   = 2.f * __logf(s) + __logf(num) - 2.f * __logf(den);
  bool outside = (x <= RMINV) || (x >= RMAXV);
  y  = outside ? x : yy;
  ld = outside ? 0.f : l;
}

__global__ __launch_bounds__(NTHREADS, 1) void flow_logprob_kernel(
    const float* __restrict__ xg,  const float* __restrict__ W0g,
    const float* __restrict__ b0g, const float* __restrict__ W1g,
    const float* __restrict__ b1g, const float* __restrict__ W2g,
    const float* __restrict__ b2g, const float* __restrict__ Woutg,
    const float* __restrict__ boutg, const float* __restrict__ sclg,
    const float* __restrict__ shfg, float* __restrict__ out, int B) {
  __shared__ SM sm;
  const int tid = threadIdx.x;
  const int wv  = tid >> 5;
  const int ln  = tid & 31;
  const int g   = ln >> 4;       // half-wave group
  const int m   = ln & 15;       // A-row / B-col / C-col / spline row
  const int rowbase = blockIdx.x * (WAVES * 16) + wv * 16;

  // ---- load this wave's 16x16 z tile -------------------------------------
  for (int idx = ln; idx < 256; idx += 32) {
    int r = idx >> 4, f = idx & 15;
    sm.zbuf[wv][r][f] = xg[(size_t)(rowbase + r) * FDIM + f];
  }
  float ldacc = 0.f;   // per lane: row m, features {g, 2+g, ..., 14+g}

  // ---- layers (reversed order) -------------------------------------------
  for (int i = NLAY - 1; i >= 0; --i) {
    __syncthreads();
    // ---- stage layer weights into LDS, transposed + bf16 + padding -------
    for (int idx = tid; idx < 256; idx += NTHREADS) {
      sm.w0t[idx & 15][idx >> 4] = (bf16_t)W0g[i * 256 + idx];
      sm.w0t[idx >> 4][16 + (idx & 15)] = (bf16_t)0.f;   // disjoint pad
    }
    for (int idx = tid; idx < 1024; idx += NTHREADS) {
      sm.w1t[idx & 63][idx >> 6] = (bf16_t)W1g[i * 1024 + idx];
      sm.w1t[idx & 63][16 + (idx >> 6)] = (bf16_t)0.f;   // disjoint pad
    }
    for (int idx = tid; idx < 4096; idx += NTHREADS)
      sm.w2t[idx & 63][idx >> 6] = (bf16_t)W2g[i * 4096 + idx];
    // Wout remap: global col n = f*25+j -> LDS row f*32+j
    for (int idx = tid; idx < 64 * POUT; idx += NTHREADS) {
      int k = idx / POUT, n = idx % POUT;
      int f = n / NBP,    j = n % NBP;
      sm.woutt[f * PPAD + j][k] = (bf16_t)Woutg[i * 64 * POUT + idx];
    }
    // zero Wout pad rows (j = 25..31), disjoint from fill
    for (int idx = tid; idx < FDIM * 7 * 64; idx += NTHREADS) {
      int f = idx / (7 * 64), rem = idx % (7 * 64);
      sm.woutt[f * PPAD + 25 + rem / 64][rem % 64] = (bf16_t)0.f;
    }
    for (int idx = tid; idx < 16; idx += NTHREADS) {
      sm.b0v[idx]  = b0g[i * 16 + idx];
      float sc = sclg[i * 16 + idx];
      sm.sclv[idx] = sc;
      sm.shfv[idx] = shfg[i * 16 + idx];
      sm.lscv[idx] = __logf(fabsf(sc));
    }
    for (int idx = tid; idx < 64; idx += NTHREADS) {
      sm.b1v[idx] = b1g[i * 64 + idx];
      sm.b2v[idx] = b2g[i * 64 + idx];
    }
    for (int idx = tid; idx < POUT; idx += NTHREADS) {
      int f = idx / NBP, j = idx % NBP;
      sm.boutv[f * PPAD + j] = boutg[i * POUT + idx];
    }
    for (int idx = tid; idx < FDIM * 7; idx += NTHREADS)
      sm.boutv[(idx / 7) * PPAD + 25 + idx % 7] = 0.f;   // disjoint pad
    // prefetch next layer's big weight block (gfx1250 global_prefetch)
    if (i > 0) {
      const float* nxt = Woutg + (size_t)(i - 1) * 64 * POUT;
      for (int c = tid * 16; c < 64 * POUT; c += NTHREADS * 16)
        __builtin_prefetch(nxt + c, 0, 1);
    }
    __syncthreads();

    // ---- GEMM 1: hid0 = tanh(mz @ W0 + b0), K=16 padded ------------------
    v16bf amz;
#pragma unroll
    for (int j = 0; j < 16; ++j) {
      int k = (j < 8) ? (8 * g + j) : (16 + 8 * g + (j - 8));
      float zv = (k < FDIM) ? sm.zbuf[wv][m][k] : 0.f;
      bool msk = (((k + i) & 1) != 0);
      amz[j] = (bf16_t)(msk ? zv : 0.f);
    }
    v8f c0 = {0.f, 0.f, 0.f, 0.f, 0.f, 0.f, 0.f, 0.f};
    c0 = wmma_bf16(amz, load_bfrag(&sm.w0t[0][0], 32, m, g, 0), c0);
#pragma unroll
    for (int v = 0; v < 8; ++v) {
      int r = 8 * g + v;
      sm.hidA[wv][r][m] = (bf16_t)tanh_fast(c0[v] + sm.b0v[m]);
      sm.hidA[wv][r][16 + m] = (bf16_t)0.f;  // K pad for next GEMM
    }
    LDS_FENCE();

    // ---- GEMM 2: hid1 = tanh(hid0 @ W1 + b1), 16x64, K=16 padded ---------
    {
      v16bf a1 = load_afrag(&sm.hidA[wv][0][0], 64, m, g, 0);
#pragma unroll
      for (int t = 0; t < 4; ++t) {
        v8f c = {0.f, 0.f, 0.f, 0.f, 0.f, 0.f, 0.f, 0.f};
        c = wmma_bf16(a1, load_bfrag(&sm.w1t[0][0], 32, t * 16 + m, g, 0), c);
        int n = t * 16 + m;
#pragma unroll
        for (int v = 0; v < 8; ++v)
          sm.hidB[wv][8 * g + v][n] = (bf16_t)tanh_fast(c[v] + sm.b1v[n]);
      }
    }
    LDS_FENCE();

    // ---- GEMM 3: hid2 = hid1 @ W2 + b2, 16x64, K=64 ----------------------
    {
      v16bf alo = load_afrag(&sm.hidB[wv][0][0], 64, m, g, 0);
      v16bf ahi = load_afrag(&sm.hidB[wv][0][0], 64, m, g, 32);
#pragma unroll
      for (int t = 0; t < 4; ++t) {
        v8f c = {0.f, 0.f, 0.f, 0.f, 0.f, 0.f, 0.f, 0.f};
        c = wmma_bf16(alo, load_bfrag(&sm.w2t[0][0], 64, t * 16 + m, g, 0), c);
        c = wmma_bf16(ahi, load_bfrag(&sm.w2t[0][0], 64, t * 16 + m, g, 32), c);
        int n = t * 16 + m;
#pragma unroll
        for (int v = 0; v < 8; ++v)
          sm.hidA[wv][8 * g + v][n] = (bf16_t)(c[v] + sm.b2v[n]);
      }
    }
    LDS_FENCE();

    // ---- GEMM 4 fused with spline: 8 chunks x (4 tiles + 2 features) -----
    {
      v16bf alo = load_afrag(&sm.hidA[wv][0][0], 64, m, g, 0);
      v16bf ahi = load_afrag(&sm.hidA[wv][0][0], 64, m, g, 32);
      for (int tg = 0; tg < 8; ++tg) {
#pragma unroll
        for (int tt = 0; tt < 4; ++tt) {
          int n = (tg * 4 + tt) * 16 + m;            // padded column
          v8f c = {0.f, 0.f, 0.f, 0.f, 0.f, 0.f, 0.f, 0.f};
          c = wmma_bf16(alo, load_bfrag(&sm.woutt[0][0], 64, n, g, 0), c);
          c = wmma_bf16(ahi, load_bfrag(&sm.woutt[0][0], 64, n, g, 32), c);
#pragma unroll
          for (int v = 0; v < 8; ++v)
            sm.pchunk[wv][8 * g + v][tt * 16 + m] = c[v] + sm.boutv[n];
        }
        LDS_FENCE();
        // spline: lane owns (row m, feature f = 2*tg + g)
        {
          const int  f   = 2 * tg + g;
          const bool msk = (((f + i) & 1) != 0);
          float x = sm.zbuf[wv][m][f];
          float y, ld;
          rq_spline(&sm.pchunk[wv][m][g * PPAD], x, y, ld);
          float zn = msk ? x : y;
          ldacc += (msk ? 0.f : ld) + sm.lscv[f];
          sm.zbuf[wv][m][f] = zn * sm.sclv[f] + sm.shfv[f];
        }
        LDS_FENCE();  // spline reads done before next chunk overwrites
      }
    }
  }

  // ---- base log-prob + reduce, write output ------------------------------
  {
    float t = ldacc;
#pragma unroll
    for (int tg = 0; tg < 8; ++tg) {
      float z = sm.zbuf[wv][m][2 * tg + g];
      t -= 0.5f * z * z;
    }
    t += __shfl_xor(t, 16, 32);   // combine even/odd feature halves
    if (ln < 16 && (rowbase + m) < B) out[rowbase + m] = t + BASE_C;
  }
}

extern "C" void kernel_launch(void* const* d_in, const int* in_sizes, int n_in,
                              void* d_out, int out_size, void* d_ws, size_t ws_size,
                              hipStream_t stream) {
  const float* x    = (const float*)d_in[0];
  const float* W0   = (const float*)d_in[1];
  const float* b0   = (const float*)d_in[2];
  const float* W1   = (const float*)d_in[3];
  const float* b1   = (const float*)d_in[4];
  const float* W2   = (const float*)d_in[5];
  const float* b2   = (const float*)d_in[6];
  const float* Wout = (const float*)d_in[7];
  const float* bout = (const float*)d_in[8];
  const float* scl  = (const float*)d_in[9];
  const float* shf  = (const float*)d_in[10];
  float* out = (float*)d_out;
  int B = in_sizes[0] / FDIM;
  int grid = (B + WAVES * 16 - 1) / (WAVES * 16);
  flow_logprob_kernel<<<grid, NTHREADS, 0, stream>>>(
      x, W0, b0, W1, b1, W2, b2, Wout, bout, scl, shf, out, B);
}
